// GRNN_26113401160017
// MI455X (gfx1250) — matrix operand order repaired and checked by reading, not compiled, not run on hardware
//
#include <hip/hip_runtime.h>
#include <hip/hip_bf16.h>

// Problem constants (fixed by the reference harness).
constexpr int B  = 8;
constexpr int S  = 32;
constexpr int N  = 2048;
constexpr int D  = 64;
constexpr int H  = 128;
constexpr int E  = 32768;
constexpr int NT = B * N;        // 16384 nodes total
constexpr int K  = D + H;        // 192 GEMM inner dim
constexpr int ETOT = B * E;      // 262144 batched edges
constexpr float LN_EPS = 1e-5f;

typedef __attribute__((ext_vector_type(16))) __bf16 v16bf;
typedef __attribute__((ext_vector_type(8)))  float  v8f;

// ---------------------------------------------------------------------------
// One-time: pack W [K x H] fp32 (row-major, rows = K) into the CDNA5 WMMA
// bf16 B-fragment layout. For each k-step ks (K=32 slice) and column tile w
// (16 cols), lane n<16 holds column 16w+n with K rows ks*32+0..7,16..23 packed
// two per dword; lanes 16..31 hold K rows ks*32+8..15,24..31. Fragment = 8 dwords/lane.
__global__ void pack_w_kernel(const float* __restrict__ W,
                              __hip_bfloat16* __restrict__ wp) {
  int t = blockIdx.x * blockDim.x + threadIdx.x;
  const int total = (K / 32) * (H / 16) * 32 * 8;
  if (t >= total) return;
  int j    = t & 7;
  int lane = (t >> 3) & 31;
  int w    = (t >> 8) & 7;
  int ks   = t >> 11;
  int col  = 16 * w + (lane & 15);
  int krow = 32 * ks + ((lane >= 16) ? 8 : 0) + 2 * j + ((j >= 4) ? 8 : 0);
  wp[2 * t + 0] = __float2bfloat16(W[(size_t)krow * H + col]);
  wp[2 * t + 1] = __float2bfloat16(W[(size_t)(krow + 1) * H + col]);
}

// ---------------------------------------------------------------------------
// Degree with self-loop, then dinv = rsqrt(deg) in place.
__global__ void deg_init_kernel(float* __restrict__ deg) {
  int i = blockIdx.x * blockDim.x + threadIdx.x;
  if (i < NT) deg[i] = 1.0f;  // self loop
}

__global__ void deg_scatter_kernel(const long long* __restrict__ ei,
                                   float* __restrict__ deg) {
  int e = blockIdx.x * blockDim.x + threadIdx.x;
  if (e >= ETOT) return;
  int b  = e / E;
  int er = e - b * E;
  int dst = (int)ei[E + er] + b * N;
  atomicAdd(&deg[dst], 1.0f);
}

__global__ void dinv_kernel(float* __restrict__ deg) {
  int i = blockIdx.x * blockDim.x + threadIdx.x;
  if (i < NT) deg[i] = rsqrtf(deg[i]);  // deg >= 1 always
}

// ---------------------------------------------------------------------------
// Build bf16 XH = [x_t | h] (NT x 192). The x part is also written into the
// candidate input matrix XRH (its h-part is filled later by gates_kernel).
__global__ void pack_xh_kernel(const float* __restrict__ xseq,
                               const float* __restrict__ h,
                               __hip_bfloat16* __restrict__ xh,
                               __hip_bfloat16* __restrict__ xrh, int s) {
  int idx = blockIdx.x * blockDim.x + threadIdx.x;
  if (idx >= NT * K) return;
  int n = idx / K;
  int c = idx - n * K;
  if (c < D) {
    int b = n / N, node = n - b * N;
    float v = xseq[(((size_t)b * S + s) * N + node) * D + c];
    __hip_bfloat16 bv = __float2bfloat16(v);
    xh[idx]  = bv;
    xrh[idx] = bv;
  } else {
    xh[idx] = __float2bfloat16(h[(size_t)n * H + (c - D)]);
  }
}

// ---------------------------------------------------------------------------
// GEMM: out[NT x H] = XH[NT x K](bf16) @ Wpacked(K x H, bf16, fragment order).
// 256 threads = 8 waves; wave w owns output columns [16w,16w+16); the block
// owns 64 rows. Each wave keeps 4 independent accumulators (4 M-subtiles),
// so each B-fragment is loaded once per K-step and feeds 4 *independent*
// back-to-back v_wmma_f32_16x16x32_bf16 ops (covers the bf16 WMMA RAW
// hazard without v_nops and cuts B-fragment loads 4x vs 16-row blocks).
__global__ void __launch_bounds__(256)
gemm_kernel(const __hip_bfloat16* __restrict__ xh,
            const __hip_bfloat16* __restrict__ wp,
            float* __restrict__ out) {
  const int m0   = blockIdx.x * 64;
  const int wave = threadIdx.x >> 5;
  const int lane = threadIdx.x & 31;
  const int rA   = lane & 15;
  const int hiA  = (lane >> 4) * 8;  // lanes 0-15: K+0..7/16..23; 16-31: +8/+24

  union AFrag { uint4 u[2]; v16bf v; };
  v8f acc0 = {}, acc1 = {}, acc2 = {}, acc3 = {};

#pragma unroll
  for (int ks = 0; ks < K / 32; ++ks) {
    v16bf bfrag = ((const v16bf*)wp)[(ks * 8 + wave) * 32 + lane];
    const __hip_bfloat16* ap = xh + (size_t)(m0 + rA) * K + ks * 32 + hiA;
    AFrag a0, a1, a2, a3;
    a0.u[0] = *(const uint4*)(ap);
    a0.u[1] = *(const uint4*)(ap + 16);
    a1.u[0] = *(const uint4*)(ap + 16 * K);
    a1.u[1] = *(const uint4*)(ap + 16 * K + 16);
    a2.u[0] = *(const uint4*)(ap + 32 * K);
    a2.u[1] = *(const uint4*)(ap + 32 * K + 16);
    a3.u[0] = *(const uint4*)(ap + 48 * K);
    a3.u[1] = *(const uint4*)(ap + 48 * K + 16);
    acc0 = __builtin_amdgcn_wmma_f32_16x16x32_bf16(false, a0.v, false, bfrag,
                                                   (short)0, acc0, false, false);
    acc1 = __builtin_amdgcn_wmma_f32_16x16x32_bf16(false, a1.v, false, bfrag,
                                                   (short)0, acc1, false, false);
    acc2 = __builtin_amdgcn_wmma_f32_16x16x32_bf16(false, a2.v, false, bfrag,
                                                   (short)0, acc2, false, false);
    acc3 = __builtin_amdgcn_wmma_f32_16x16x32_bf16(false, a3.v, false, bfrag,
                                                   (short)0, acc3, false, false);
  }

  // C/D layout: VGPR j -> row base + j (+8 for lanes 16-31), col = lane&15.
  const int col   = wave * 16 + (lane & 15);
  const int rhalf = (lane >> 4) ? 8 : 0;
  float* o0 = out + (size_t)(m0 +  0 + rhalf) * H + col;
  float* o1 = out + (size_t)(m0 + 16 + rhalf) * H + col;
  float* o2 = out + (size_t)(m0 + 32 + rhalf) * H + col;
  float* o3 = out + (size_t)(m0 + 48 + rhalf) * H + col;
#pragma unroll
  for (int j = 0; j < 8; ++j) {
    o0[(size_t)j * H] = acc0[j];
    o1[(size_t)j * H] = acc1[j];
    o2[(size_t)j * H] = acc2[j];
    o3[(size_t)j * H] = acc3[j];
  }
}

// ---------------------------------------------------------------------------
// A[n][c] = bias[c] + dinv[n]^2 * G[n][c]   (bias + self-loop contribution)
__global__ void init_agg_kernel(const float* __restrict__ g,
                                const float* __restrict__ dinv,
                                const float* __restrict__ bias,
                                float* __restrict__ a) {
  int idx = blockIdx.x * blockDim.x + threadIdx.x;
  if (idx >= NT * H) return;
  int n = idx >> 7;
  int c = idx & (H - 1);
  float dv = dinv[n];
  a[idx] = bias[c] + dv * dv * g[idx];
}

// Edge scatter: A[dst] += dinv[src]*dinv[dst] * G[src]; one thread per
// (edge, float4-chunk of H).
__global__ void scatter_kernel(const long long* __restrict__ ei,
                               const float* __restrict__ dinv,
                               const float* __restrict__ g,
                               float* __restrict__ a) {
  int t = blockIdx.x * blockDim.x + threadIdx.x;
  if (t >= ETOT * (H / 4)) return;
  int c = t & (H / 4 - 1);
  int e = t >> 5;
  int b  = e / E;
  int er = e - b * E;
  int src = (int)ei[er] + b * N;
  int dst = (int)ei[E + er] + b * N;
  float w = dinv[src] * dinv[dst];
  float4 gv = ((const float4*)g)[(size_t)src * (H / 4) + c];
  float* ap = a + (size_t)dst * H + c * 4;
  atomicAdd(ap + 0, w * gv.x);
  atomicAdd(ap + 1, w * gv.y);
  atomicAdd(ap + 2, w * gv.z);
  atomicAdd(ap + 3, w * gv.w);
}

// ---------------------------------------------------------------------------
// z = sigmoid(Az) in place; r = sigmoid(Ar); XRH h-part = bf16(r * h).
__global__ void gates_kernel(float* __restrict__ az,
                             const float* __restrict__ ar,
                             const float* __restrict__ h,
                             __hip_bfloat16* __restrict__ xrh) {
  int idx = blockIdx.x * blockDim.x + threadIdx.x;
  if (idx >= NT * H) return;
  int n = idx >> 7;
  int c = idx & (H - 1);
  float z = 1.0f / (1.0f + expf(-az[idx]));
  float r = 1.0f / (1.0f + expf(-ar[idx]));
  az[idx] = z;
  xrh[(size_t)n * K + D + c] = __float2bfloat16(r * h[idx]);
}

// ---------------------------------------------------------------------------
// h = LayerNorm((1-z)*h + z*tanh(Ac)) — one wave32 per node, 4 elems/lane.
__global__ void __launch_bounds__(256)
update_ln_kernel(const float* __restrict__ ac, const float* __restrict__ z,
                 const float* __restrict__ gamma, const float* __restrict__ beta,
                 float* __restrict__ h) {
  int node = blockIdx.x * 8 + (threadIdx.x >> 5);
  int lane = threadIdx.x & 31;
  const size_t base = (size_t)node * (H / 4);
  float4 hv = ((const float4*)h)[base + lane];
  float4 zv = ((const float4*)z)[base + lane];
  float4 av = ((const float4*)ac)[base + lane];
  float4 hn;
  hn.x = (1.0f - zv.x) * hv.x + zv.x * tanhf(av.x);
  hn.y = (1.0f - zv.y) * hv.y + zv.y * tanhf(av.y);
  hn.z = (1.0f - zv.z) * hv.z + zv.z * tanhf(av.z);
  hn.w = (1.0f - zv.w) * hv.w + zv.w * tanhf(av.w);

  float s = hn.x + hn.y + hn.z + hn.w;
#pragma unroll
  for (int m = 16; m >= 1; m >>= 1) s += __shfl_xor(s, m, 32);
  float mu = s * (1.0f / H);

  float dx = hn.x - mu, dy = hn.y - mu, dz = hn.z - mu, dw = hn.w - mu;
  float q = dx * dx + dy * dy + dz * dz + dw * dw;
#pragma unroll
  for (int m = 16; m >= 1; m >>= 1) q += __shfl_xor(q, m, 32);
  float inv = rsqrtf(q * (1.0f / H) + LN_EPS);

  float4 g4 = ((const float4*)gamma)[lane];
  float4 b4 = ((const float4*)beta)[lane];
  float4 o;
  o.x = dx * inv * g4.x + b4.x;
  o.y = dy * inv * g4.y + b4.y;
  o.z = dz * inv * g4.z + b4.z;
  o.w = dw * inv * g4.w + b4.w;
  ((float4*)h)[base + lane] = o;
}

// ---------------------------------------------------------------------------
// out[b][c] = mean over nodes of h — coalesced per-column sums.
__global__ void pool_kernel(const float* __restrict__ h, float* __restrict__ out) {
  int b = blockIdx.x;
  int c = threadIdx.x;  // blockDim = H
  float s = 0.0f;
  for (int i = 0; i < N; ++i) s += h[((size_t)(b * N + i)) * H + c];
  out[b * H + c] = s * (1.0f / N);
}

// ---------------------------------------------------------------------------
extern "C" void kernel_launch(void* const* d_in, const int* in_sizes, int n_in,
                              void* d_out, int out_size, void* d_ws, size_t ws_size,
                              hipStream_t stream) {
  (void)in_sizes; (void)n_in; (void)out_size; (void)ws_size;
  const float*     xseq  = (const float*)d_in[0];
  const long long* ei    = (const long long*)d_in[1];
  // d_in[2] (batch) unused: reference pools contiguous N-blocks.
  const float* Wz = (const float*)d_in[3];
  const float* bz = (const float*)d_in[4];
  const float* Wr = (const float*)d_in[5];
  const float* br = (const float*)d_in[6];
  const float* Wc = (const float*)d_in[7];
  const float* bc = (const float*)d_in[8];
  const float* gamma = (const float*)d_in[9];
  const float* beta  = (const float*)d_in[10];
  float* out = (float*)d_out;

  size_t off = 0;
  auto alloc = [&](size_t bytes) {
    void* p = (char*)d_ws + off;
    off += (bytes + 255) & ~(size_t)255;
    return p;
  };
  float*          h    = (float*)alloc((size_t)NT * H * 4);
  __hip_bfloat16* xh   = (__hip_bfloat16*)alloc((size_t)NT * K * 2);
  __hip_bfloat16* xrh  = (__hip_bfloat16*)alloc((size_t)NT * K * 2);
  __hip_bfloat16* wzp  = (__hip_bfloat16*)alloc((size_t)K * H * 2);
  __hip_bfloat16* wrp  = (__hip_bfloat16*)alloc((size_t)K * H * 2);
  __hip_bfloat16* wcp  = (__hip_bfloat16*)alloc((size_t)K * H * 2);
  float*          dinv = (float*)alloc((size_t)NT * 4);
  float*          gz   = (float*)alloc((size_t)NT * H * 4);  // also reused as Gc
  float*          gr   = (float*)alloc((size_t)NT * H * 4);  // also reused as Ac
  float*          az   = (float*)alloc((size_t)NT * H * 4);  // becomes z in place
  float*          ar   = (float*)alloc((size_t)NT * H * 4);

  hipMemsetAsync(h, 0, (size_t)NT * H * 4, stream);  // h0 = zeros, every call

  const int packw_blocks = ((K / 32) * (H / 16) * 32 * 8 + 255) / 256;  // 48
  pack_w_kernel<<<packw_blocks, 256, 0, stream>>>(Wz, wzp);
  pack_w_kernel<<<packw_blocks, 256, 0, stream>>>(Wr, wrp);
  pack_w_kernel<<<packw_blocks, 256, 0, stream>>>(Wc, wcp);

  deg_init_kernel<<<NT / 256, 256, 0, stream>>>(dinv);
  deg_scatter_kernel<<<ETOT / 256, 256, 0, stream>>>(ei, dinv);
  dinv_kernel<<<NT / 256, 256, 0, stream>>>(dinv);

  const int ew_blocks  = (NT * H) / 256;           // 8192 elementwise blocks
  const int sc_blocks  = (ETOT * (H / 4)) / 256;   // 32768 scatter blocks
  const int gemm_grid  = NT / 64;                  // 256 blocks, 64 rows each

  for (int s = 0; s < S; ++s) {
    pack_xh_kernel<<<(NT * K) / 256, 256, 0, stream>>>(xseq, h, xh, xrh, s);
    // z / r gates
    gemm_kernel<<<gemm_grid, 256, 0, stream>>>(xh, wzp, gz);
    gemm_kernel<<<gemm_grid, 256, 0, stream>>>(xh, wrp, gr);
    init_agg_kernel<<<ew_blocks, 256, 0, stream>>>(gz, dinv, bz, az);
    init_agg_kernel<<<ew_blocks, 256, 0, stream>>>(gr, dinv, br, ar);
    scatter_kernel<<<sc_blocks, 256, 0, stream>>>(ei, dinv, gz, az);
    scatter_kernel<<<sc_blocks, 256, 0, stream>>>(ei, dinv, gr, ar);
    gates_kernel<<<ew_blocks, 256, 0, stream>>>(az, ar, h, xrh);
    // candidate
    gemm_kernel<<<gemm_grid, 256, 0, stream>>>(xrh, wcp, gz);          // Gc -> gz
    init_agg_kernel<<<ew_blocks, 256, 0, stream>>>(gz, dinv, bc, gr);  // Ac -> gr
    scatter_kernel<<<sc_blocks, 256, 0, stream>>>(ei, dinv, gz, gr);
    // blend + layernorm (az now holds z)
    update_ln_kernel<<<NT / 8, 256, 0, stream>>>(gr, az, gamma, beta, h);
  }

  pool_kernel<<<B, H, 0, stream>>>(h, out);
}
